// TreeEnergyLoss_6708738916948
// MI455X (gfx1250) — compile-verified, matrix operand order checked by reading, not run on hardware
//
#include <hip/hip_runtime.h>
#include <hip/hip_bf16.h>

typedef __attribute__((ext_vector_type(16))) _Float16 v16h;
typedef __attribute__((ext_vector_type(8)))  _Float16 v8h;
typedef __attribute__((ext_vector_type(8)))  float    v8f;

#define BN     8
#define NCLS   21
#define HH     96
#define WW     96
#define VV     (HH*WW)                       // 9216
#define H0C    384
#define W0C    384
#define EE     ((HH-1)*WW + HH*(WW-1))       // 18240
#define CHI    256
#define FC     (NCLS+1)                      // 22
#define SIGMA_F 2.0f
#define LWT    0.4f
#define SORT_N 32768
#define TILE_N 2048
#define LSS    (VV+2)

__device__ __forceinline__ void edge_uv(int e, int& u, int& v) {
  const int EVERT = (HH-1)*WW;
  if (e < EVERT) { u = e; v = e + WW; }
  else { int k = e - EVERT; int r = k/(WW-1); int c = k - r*(WW-1); u = r*WW + c; v = u + 1; }
}

// ---------------------------------------------------------------- prep
__global__ void tel_init_edges(int* __restrict__ eu, int* __restrict__ ev) {
  int e = blockIdx.x*blockDim.x + threadIdx.x;
  if (e >= EE) return;
  int u, v; edge_uv(e, u, v);
  eu[e] = u; ev[e] = v;
}

__global__ void tel_prep(const float* __restrict__ preds, const float* __restrict__ low,
                         const int* __restrict__ roiI,
                         float* __restrict__ prob, float* __restrict__ lowr, float* __restrict__ roi)
{
  int i = blockIdx.x*blockDim.x + threadIdx.x;
  if (i >= BN*VV) return;
  int b = i / VV, p = i - b*VV;
  int y = p / WW, x = p - y*WW;
  // softmax over classes
  const float* pr = preds + ((size_t)b*NCLS)*VV + p;
  float mx = -3.4e38f;
#pragma unroll
  for (int c = 0; c < NCLS; c++) mx = fmaxf(mx, pr[(size_t)c*VV]);
  float s = 0.f, ex[NCLS];
#pragma unroll
  for (int c = 0; c < NCLS; c++) { ex[c] = __expf(pr[(size_t)c*VV] - mx); s += ex[c]; }
  float inv = 1.f/s;
#pragma unroll
  for (int c = 0; c < NCLS; c++) prob[((size_t)b*NCLS + c)*VV + p] = ex[c]*inv;
  // anti-aliased triangle (jax bilinear) 384 -> 96 downsample, half-pixel centers
  float cy = (y + 0.5f)*4.f, cx = (x + 0.5f)*4.f;
  float a0 = 0.f, a1 = 0.f, a2 = 0.f, wsum = 0.f;
  for (int sy = 4*y - 2; sy <= 4*y + 5; sy++) {
    float wy = 1.f - fabsf(((float)sy + 0.5f) - cy)*0.25f;
    if (wy <= 0.f) continue;
    int yy = sy < 0 ? 0 : (sy > H0C-1 ? H0C-1 : sy);
    for (int sx = 4*x - 2; sx <= 4*x + 5; sx++) {
      float wx = 1.f - fabsf(((float)sx + 0.5f) - cx)*0.25f;
      if (wx <= 0.f) continue;
      int xx = sx < 0 ? 0 : (sx > W0C-1 ? W0C-1 : sx);
      float w = wy*wx; wsum += w;
      size_t base = ((size_t)b*3)*H0C*W0C + (size_t)yy*W0C + xx;
      a0 += w*low[base];
      a1 += w*low[base + (size_t)H0C*W0C];
      a2 += w*low[base + 2*(size_t)H0C*W0C];
    }
  }
  float iw = 1.f/wsum;
  lowr[((size_t)b*3 + 0)*VV + p] = a0*iw;
  lowr[((size_t)b*3 + 1)*VV + p] = a1*iw;
  lowr[((size_t)b*3 + 2)*VV + p] = a2*iw;
  roi[i] = (float)roiI[(size_t)b*H0C*W0C + (size_t)(y*4)*W0C + (x*4)];
}

// -------------------------------------- transpose + f16 pack high_feats
// [B][256][V] f32  ->  [B][V][256] f16, 32x32 LDS tiles (conflict-padded)
__global__ __launch_bounds__(256)
void tel_pack_f16(const float* __restrict__ hf, _Float16* __restrict__ hfT)
{
  __shared__ _Float16 sd[32][33];
  int b = blockIdx.z;
  int v0 = blockIdx.x*32, c0 = blockIdx.y*32;
  const float* src = hf + ((size_t)b*CHI)*VV;
#pragma unroll
  for (int r = threadIdx.y; r < 32; r += 8)
    sd[r][threadIdx.x] = (_Float16)src[(size_t)(c0 + r)*VV + v0 + threadIdx.x];
  __syncthreads();
  _Float16* dst = hfT + (size_t)b*VV*CHI;
#pragma unroll
  for (int r = threadIdx.y; r < 32; r += 8)
    dst[(size_t)(v0 + r)*CHI + c0 + threadIdx.x] = sd[threadIdx.x][r];
}

// ---------------------------------------------------- pairwise distances
// 3-channel squared distance (low embedding): plain VALU.
__global__ void tel_dist3(const float* __restrict__ emb, size_t embBS,
                          const int* __restrict__ ia, int iaBS,
                          const int* __restrict__ ib, int ibBS,
                          float* __restrict__ o, int oBS, int n)
{
  int b = blockIdx.y;
  int i = blockIdx.x*blockDim.x + threadIdx.x;
  if (i >= n) return;
  const float* eb = emb + (size_t)b*embBS;
  int va = ia[(size_t)b*iaBS + i], vb = ib[(size_t)b*ibBS + i];
  float s = 0.f;
#pragma unroll
  for (int c = 0; c < 3; c++) {
    float d = eb[(size_t)c*VV + va] - eb[(size_t)c*VV + vb];
    s += d*d;
  }
  o[(size_t)b*oBS + i] = s;
}

// 256-channel squared distance via WMMA row-sum-with-ones on packed f16 rows.
// Lane's two K-runs (8 halves each) are contiguous 16B in [V][256] layout ->
// two global_load_b128 per chunk per vertex.  8 chained wmma cover 256 ch.
__global__ __launch_bounds__(256)
void tel_dist256_wmma(const _Float16* __restrict__ hfT, size_t bStride,
                      const int* __restrict__ ia, int iaBS,
                      const int* __restrict__ ib, int ibBS,
                      float* __restrict__ o, int oBS, int n)
{
  int b = blockIdx.y;
  const _Float16* base = hfT + (size_t)b*bStride;
  const int* A  = ia + (size_t)b*iaBS;
  const int* Bp = ib + (size_t)b*ibBS;
  int lane = threadIdx.x & 31;
  int wave = blockIdx.x*(blockDim.x >> 5) + (threadIdx.x >> 5);
  int m0 = wave*16;
  if (m0 >= n) return;                 // wave-uniform: EXEC stays full for WMMA
  int mrow = lane & 15;
  int p = m0 + mrow; if (p >= n) p = n - 1;   // clamp keeps all lanes active
  int va = A[p], vb = Bp[p];
  const v8h* pa = (const v8h*)(base + (size_t)va*CHI);
  const v8h* pb = (const v8h*)(base + (size_t)vb*CHI);
  int koff = (lane >= 16) ? 1 : 0;     // ISA 16-bit A-matrix lane split (+8 halves)
  v16h ones;
#pragma unroll
  for (int t = 0; t < 16; t++) ones[t] = (_Float16)1.0f;
  v8f acc = {};
#pragma unroll
  for (int q = 0; q < 8; q++) {
    int o0 = 4*q + koff;               // halves (32q + khi)/8
    if (q < 7) __builtin_prefetch((const void*)(pa + o0 + 4), 0, 1);
    v8h a0 = pa[o0],     b0 = pb[o0];      // K-run {khi .. khi+7}
    v8h a1 = pa[o0 + 2], b1 = pb[o0 + 2];  // K-run {khi+16 .. khi+23}
    v8h d0 = a0 - b0, d1 = a1 - b1;
    v8h s0 = d0*d0,   s1 = d1*d1;
    v16h a;
#pragma unroll
    for (int t = 0; t < 8; t++) { a[t] = s0[t]; a[t + 8] = s1[t]; }
    acc = __builtin_amdgcn_wmma_f32_16x16x32_f16(false, a, false, ones,
                                                 (short)0, acc, false, false);
  }
  // C/D layout: lane 0 holds rows 0..7 of column 0, lane 16 rows 8..15.
  if (lane == 0) {
#pragma unroll
    for (int r = 0; r < 8; r++) if (m0 + r < n) o[(size_t)b*oBS + m0 + r] = acc[r];
  } else if (lane == 16) {
#pragma unroll
    for (int r = 0; r < 8; r++) if (m0 + 8 + r < n) o[(size_t)b*oBS + m0 + 8 + r] = acc[r];
  }
}

// ------------------------------------------------------------- edge sort
// Per-batch bitonic sort of 32768 u64 keys (weight bits << 32 | idx):
// tiles of 2048 sorted in LDS, cross-tile strides in global, tails in LDS.
__global__ __launch_bounds__(1024)
void tel_sort(const float* __restrict__ edgew, unsigned long long* __restrict__ gk,
              int* __restrict__ sortedIdx)
{
  __shared__ unsigned long long sk[TILE_N];
  int b = blockIdx.x;
  unsigned long long* K = gk + (size_t)b*SORT_N;
  for (int i = threadIdx.x; i < SORT_N; i += 1024) {
    unsigned long long k = 0xFFFFFFFFFFFFFFFFull;
    if (i < EE) {
      unsigned bits = __float_as_uint(edgew[(size_t)b*EE + i]);  // weights >= 0
      k = ((unsigned long long)bits << 32) | (unsigned)i;
    }
    K[i] = k;
  }
  __syncthreads();
  // phase 1: sort each tile fully in LDS (sizes 2..TILE_N)
  for (int tile = 0; tile < SORT_N/TILE_N; tile++) {
    unsigned long long* base = K + (size_t)tile*TILE_N;
    for (int i = threadIdx.x; i < TILE_N; i += 1024) sk[i] = base[i];
    __syncthreads();
    for (int size = 2; size <= TILE_N; size <<= 1) {
      for (int stride = size >> 1; stride > 0; stride >>= 1) {
        int t = threadIdx.x;                      // TILE_N/2 == 1024 pairs
        int i = ((t/stride)*(stride << 1)) + (t % stride);
        int gi = tile*TILE_N + i;
        bool up = ((gi & size) == 0);
        unsigned long long x = sk[i], y = sk[i + stride];
        if ((x > y) == up) { sk[i] = y; sk[i + stride] = x; }
        __syncthreads();
      }
    }
    for (int i = threadIdx.x; i < TILE_N; i += 1024) base[i] = sk[i];
    __syncthreads();
  }
  // phase 2: sizes 2*TILE_N .. SORT_N
  for (int size = TILE_N << 1; size <= SORT_N; size <<= 1) {
    for (int stride = size >> 1; stride >= TILE_N; stride >>= 1) {
      for (int t = threadIdx.x; t < SORT_N/2; t += 1024) {
        int i = ((t/stride)*(stride << 1)) + (t % stride);
        bool up = ((i & size) == 0);
        unsigned long long x = K[i], y = K[i + stride];
        if ((x > y) == up) { K[i] = y; K[i + stride] = x; }
      }
      __syncthreads();
    }
    for (int tile = 0; tile < SORT_N/TILE_N; tile++) {
      unsigned long long* base = K + (size_t)tile*TILE_N;
      for (int i = threadIdx.x; i < TILE_N; i += 1024) sk[i] = base[i];
      __syncthreads();
      for (int stride = TILE_N >> 1; stride > 0; stride >>= 1) {
        int t = threadIdx.x;
        int i = ((t/stride)*(stride << 1)) + (t % stride);
        int gi = tile*TILE_N + i;
        bool up = ((gi & size) == 0);
        unsigned long long x = sk[i], y = sk[i + stride];
        if ((x > y) == up) { sk[i] = y; sk[i + stride] = x; }
        __syncthreads();
      }
      for (int i = threadIdx.x; i < TILE_N; i += 1024) base[i] = sk[i];
      __syncthreads();
    }
  }
  for (int i = threadIdx.x; i < EE; i += 1024)
    sortedIdx[(size_t)b*EE + i] = (int)(K[i] & 0xFFFFFFFFu);
}

// ------------------------------------------- Kruskal + BFS tree rooting
// Union-find lives in LDS: the path-halving dependent-load chain is the
// serial bottleneck, so its latency drops to LDS.  Parallel fills use all
// 256 threads; only the inherently serial parts run on thread 0.
__global__ __launch_bounds__(256)
void tel_kruskal_bfs(const int* __restrict__ sortedIdx,
                     int* head, int* nxt, int* adjv,
                     int* order, int* parv, int* depthv, int* rankv,
                     int* parRank, int* parVert, int* levelStart, int* numLevels)
{
  __shared__ int U[VV];                 // 36 KB: union-find, then visited flags
  int b = blockIdx.x;
  int* Hd = head + (size_t)b*VV;
  int* Nx = nxt  + (size_t)b*2*VV;
  int* Av = adjv + (size_t)b*2*VV;
  for (int i = threadIdx.x; i < VV; i += blockDim.x) { U[i] = i; Hd[i] = -1; }
  __syncthreads();
  if (threadIdx.x == 0) {
    const int* S = sortedIdx + (size_t)b*EE;
    int ne = 0;
    for (int si = 0; si < EE; si++) {
      int e = S[si];
      int u, v; edge_uv(e, u, v);
      int ra = u; while (U[ra] != ra) { U[ra] = U[U[ra]]; ra = U[ra]; }
      int rb = v; while (U[rb] != rb) { U[rb] = U[U[rb]]; rb = U[rb]; }
      if (ra != rb) {
        U[ra] = rb;
        Av[ne] = v; Nx[ne] = Hd[u]; Hd[u] = ne; ne++;
        Av[ne] = u; Nx[ne] = Hd[v]; Hd[v] = ne; ne++;
      }
    }
  }
  __syncthreads();
  for (int i = threadIdx.x; i < VV; i += blockDim.x) U[i] = 0;  // visited
  __syncthreads();
  int* Or = order  + (size_t)b*VV;
  int* Pv = parv   + (size_t)b*VV;
  int* Dp = depthv + (size_t)b*VV;
  int* Rk = rankv  + (size_t)b*VV;
  int* Ls = levelStart + (size_t)b*LSS;
  if (threadIdx.x == 0) {
    U[0] = 1; Pv[0] = 0; Dp[0] = 0; Or[0] = 0;
    int cnt = 1;
    for (int pos = 0; pos < cnt && pos < VV; pos++) {
      int x = Or[pos];
      for (int ei = Hd[x]; ei != -1; ei = Nx[ei]) {
        int y = Av[ei];
        if (!U[y]) { U[y] = 1; Pv[y] = x; Dp[y] = Dp[x] + 1; Or[cnt++] = y; }
      }
    }
    int L = 0; Ls[0] = 0;
    int dprev = 0;
    for (int k = 1; k < VV; k++) {
      int d = Dp[Or[k]];
      if (d != dprev) { Ls[++L] = k; dprev = d; }
    }
    Ls[L+1] = VV;
    numLevels[b] = L + 1;
  }
  __syncthreads();
  for (int k = threadIdx.x; k < VV; k += blockDim.x) Rk[Or[k]] = k;
  __syncthreads();
  int* Pr = parRank + (size_t)b*VV;
  int* Pw = parVert + (size_t)b*VV;
  for (int k = threadIdx.x; k < VV; k += blockDim.x) {
    int pvtx = Pv[Or[k]];
    Pw[k] = pvtx;
    Pr[k] = Rk[pvtx];
  }
}

// ------------------------------------------------------- filter pieces
__global__ void tel_ew(const float* __restrict__ dist, float* __restrict__ ewv) {
  int i = blockIdx.x*blockDim.x + threadIdx.x;
  if (i >= BN*VV) return;
  int k = i % VV;
  ewv[i] = (k == 0) ? 0.f : __expf(-dist[i]*(1.f/SIGMA_F));
}

__global__ void tel_gather(const float* __restrict__ f, const int* __restrict__ order,
                           float* __restrict__ fs)
{
  int i = blockIdx.x*blockDim.x + threadIdx.x;
  if (i >= BN*VV*FC) return;
  int c = i % FC;
  int k = (i / FC) % VV;
  int b = i / (FC*VV);
  int vtx = order[(size_t)b*VV + k];
  fs[i] = (c == NCLS) ? 1.0f : f[((size_t)b*NCLS + c)*VV + vtx];
}

// level-synchronous up (leaves->root) and down (root->leaves) scans
__global__ __launch_bounds__(512)
void tel_scan(const float* __restrict__ ewv, const int* __restrict__ parRank,
              const int* __restrict__ levelStart, const int* __restrict__ numLevels,
              float* __restrict__ agg, float* __restrict__ outb)
{
  int b = blockIdx.x;
  const float* w  = ewv + (size_t)b*VV;
  const int*   pr = parRank + (size_t)b*VV;
  const int*   ls = levelStart + (size_t)b*LSS;
  int L = numLevels[b];
  float* ag = agg  + (size_t)b*VV*FC;
  float* ot = outb + (size_t)b*VV*FC;
  for (int lvl = L - 1; lvl >= 1; lvl--) {
    int s = ls[lvl], e = ls[lvl+1];
    int cnt = (e - s)*FC;
    for (int t = threadIdx.x; t < cnt; t += blockDim.x) {
      int k = s + t/FC, c = t - (t/FC)*FC;
      atomicAdd(&ag[(size_t)pr[k]*FC + c], w[k]*ag[(size_t)k*FC + c]);
    }
    __syncthreads();
  }
  for (int t = threadIdx.x; t < FC; t += blockDim.x) ot[t] = ag[t];
  __syncthreads();
  for (int lvl = 1; lvl < L; lvl++) {
    int s = ls[lvl], e = ls[lvl+1];
    int cnt = (e - s)*FC;
    for (int t = threadIdx.x; t < cnt; t += blockDim.x) {
      int k = s + t/FC, c = t - (t/FC)*FC;
      float wk = w[k];
      ot[(size_t)k*FC + c] = wk*ot[(size_t)pr[k]*FC + c] + (1.f - wk*wk)*ag[(size_t)k*FC + c];
    }
    __syncthreads();
  }
}

__global__ void tel_scatter(const float* __restrict__ outb, const int* __restrict__ order,
                            float* __restrict__ AS)
{
  int i = blockIdx.x*blockDim.x + threadIdx.x;
  if (i >= BN*VV*NCLS) return;
  int c = i % NCLS;
  int k = (i / NCLS) % VV;
  int b = i / (NCLS*VV);
  int vtx = order[(size_t)b*VV + k];
  const float* row = outb + ((size_t)b*VV + k)*FC;
  AS[((size_t)b*NCLS + c)*VV + vtx] = row[c]/row[NCLS];
}

// --------------------------------------------------------------- loss
__global__ void tel_zero(float* acc) { if (threadIdx.x < 2 && blockIdx.x == 0) acc[threadIdx.x] = 0.f; }

__global__ void tel_loss(const float* __restrict__ prob, const float* __restrict__ AS,
                         const float* __restrict__ roi, float* __restrict__ acc)
{
  int i = blockIdx.x*blockDim.x + threadIdx.x;
  float l = 0.f, nn = 0.f;
  if (i < BN*NCLS*VV) {
    int v = i % VV;
    int b = i / (NCLS*VV);
    float r = roi[(size_t)b*VV + v];
    l = r*fabsf(prob[i] - AS[i]);
  }
  if (i < BN*VV) nn = roi[i];
#pragma unroll
  for (int off = 16; off > 0; off >>= 1) {   // wave32 reduction
    l  += __shfl_down(l, off, 32);
    nn += __shfl_down(nn, off, 32);
  }
  if ((threadIdx.x & 31) == 0) {
    if (l  != 0.f) atomicAdd(&acc[0], l);
    if (nn != 0.f) atomicAdd(&acc[1], nn);
  }
}

__global__ void tel_fin(const float* __restrict__ acc, float* __restrict__ out) {
  if (threadIdx.x == 0 && blockIdx.x == 0) {
    float L = acc[0], N = acc[1];
    float r = (N > 0.f) ? L/fmaxf(N, 1.f) : L;
    out[0] = LWT*r;
  }
}

// ---------------------------------------------------------------- host
static inline char* ws_take(char*& cur, size_t bytes) {
  char* p = cur;
  cur += (bytes + 255) & ~(size_t)255;
  return p;
}

extern "C" void kernel_launch(void* const* d_in, const int* in_sizes, int n_in,
                              void* d_out, int out_size, void* d_ws, size_t ws_size,
                              hipStream_t stream)
{
  (void)in_sizes; (void)n_in; (void)out_size; (void)ws_size;
  const float* preds = (const float*)d_in[0];
  const float* low   = (const float*)d_in[1];
  const float* high  = (const float*)d_in[2];
  const int*   roiI  = (const int*)d_in[3];
  float* out = (float*)d_out;

  char* cur = (char*)d_ws;
  float* lowr = (float*)ws_take(cur, (size_t)BN*3*VV*4);
  float* prob = (float*)ws_take(cur, (size_t)BN*NCLS*VV*4);
  float* roi  = (float*)ws_take(cur, (size_t)BN*VV*4);
  float* AS1  = (float*)ws_take(cur, (size_t)BN*NCLS*VV*4);
  float* AS2  = (float*)ws_take(cur, (size_t)BN*NCLS*VV*4);
  _Float16* hfT = (_Float16*)ws_take(cur, (size_t)BN*VV*CHI*2);
  int*   eu   = (int*)ws_take(cur, (size_t)EE*4);
  int*   ev   = (int*)ws_take(cur, (size_t)EE*4);
  float* edgew = (float*)ws_take(cur, (size_t)BN*EE*4);
  unsigned long long* keys = (unsigned long long*)ws_take(cur, (size_t)BN*SORT_N*8);
  int* sortedIdx = (int*)ws_take(cur, (size_t)BN*EE*4);
  int* head  = (int*)ws_take(cur, (size_t)BN*VV*4);
  int* nxt   = (int*)ws_take(cur, (size_t)BN*2*VV*4);
  int* adjv  = (int*)ws_take(cur, (size_t)BN*2*VV*4);
  int* order = (int*)ws_take(cur, (size_t)BN*VV*4);
  int* parv  = (int*)ws_take(cur, (size_t)BN*VV*4);
  int* depthv= (int*)ws_take(cur, (size_t)BN*VV*4);
  int* rankv = (int*)ws_take(cur, (size_t)BN*VV*4);
  int* parRank = (int*)ws_take(cur, (size_t)BN*VV*4);
  int* parVert = (int*)ws_take(cur, (size_t)BN*VV*4);
  int* levelStart = (int*)ws_take(cur, (size_t)BN*LSS*4);
  int* numLevels  = (int*)ws_take(cur, (size_t)BN*4);
  float* dist = (float*)ws_take(cur, (size_t)BN*VV*4);
  float* ewv  = (float*)ws_take(cur, (size_t)BN*VV*4);
  float* fs   = (float*)ws_take(cur, (size_t)BN*VV*FC*4);
  float* outb = (float*)ws_take(cur, (size_t)BN*VV*FC*4);
  float* acc  = (float*)ws_take(cur, 2*4);

  tel_init_edges<<<(EE + 255)/256, 256, 0, stream>>>(eu, ev);
  tel_prep<<<(BN*VV + 255)/256, 256, 0, stream>>>(preds, low, roiI, prob, lowr, roi);
  tel_pack_f16<<<dim3(VV/32, CHI/32, BN), dim3(32, 8), 0, stream>>>(high, hfT);

  for (int pass = 0; pass < 2; pass++) {
    const float* fin = (pass == 0) ? prob : AS1;
    float* ASo       = (pass == 0) ? AS1 : AS2;

    if (pass == 0)
      tel_dist3<<<dim3((EE + 255)/256, BN), 256, 0, stream>>>(lowr, (size_t)3*VV, eu, 0, ev, 0, edgew, EE, EE);
    else
      tel_dist256_wmma<<<dim3((EE + 127)/128, BN), 256, 0, stream>>>(hfT, (size_t)VV*CHI, eu, 0, ev, 0, edgew, EE, EE);

    tel_sort<<<BN, 1024, 0, stream>>>(edgew, keys, sortedIdx);
    tel_kruskal_bfs<<<BN, 256, 0, stream>>>(sortedIdx, head, nxt, adjv, order, parv,
                                            depthv, rankv, parRank, parVert, levelStart, numLevels);

    if (pass == 0)
      tel_dist3<<<dim3((VV + 255)/256, BN), 256, 0, stream>>>(lowr, (size_t)3*VV, order, VV, parVert, VV, dist, VV, VV);
    else
      tel_dist256_wmma<<<dim3((VV + 127)/128, BN), 256, 0, stream>>>(hfT, (size_t)VV*CHI, order, VV, parVert, VV, dist, VV, VV);

    tel_ew<<<(BN*VV + 255)/256, 256, 0, stream>>>(dist, ewv);
    tel_gather<<<(BN*VV*FC + 255)/256, 256, 0, stream>>>(fin, order, fs);
    tel_scan<<<BN, 512, 0, stream>>>(ewv, parRank, levelStart, numLevels, fs, outb);
    tel_scatter<<<(BN*VV*NCLS + 255)/256, 256, 0, stream>>>(outb, order, ASo);
  }

  tel_zero<<<1, 32, 0, stream>>>(acc);
  tel_loss<<<(BN*NCLS*VV + 255)/256, 256, 0, stream>>>(prob, AS2, roi, acc);
  tel_fin<<<1, 32, 0, stream>>>(acc, out);
}
